// Block_71949292143252
// MI455X (gfx1250) — matrix operand order repaired and checked by reading, not compiled, not run on hardware
//
#include <hip/hip_runtime.h>
#include <hip/hip_bf16.h>
#include <math.h>

// ---------------------------------------------------------------------------
// Transformer block for MI455X (gfx1250, wave32, WMMA).
// B=4, T=2048, D=1024, H=16, HS=64. All matmuls via v_wmma_f32_16x16x32_bf16.
// Global->LDS staging uses CDNA5 async-LDS loads (ASYNCcnt) with
// double-buffered LDS tiles so DMA overlaps WMMA.
// ---------------------------------------------------------------------------

typedef __bf16  v16bf __attribute__((ext_vector_type(16)));
typedef float   v8f   __attribute__((ext_vector_type(8)));

union FragBF { v16bf v; uint4 u[2]; };

static constexpr int Bc  = 4;
static constexpr int Tc  = 2048;
static constexpr int Dc  = 1024;
static constexpr int Hc  = 16;
static constexpr int HSc = 64;
static constexpr int Mrows = Bc * Tc;          // 8192

// ---- CDNA5 async global->LDS path (guarded; falls back to VGPR staging) ----
#if defined(__HIP_DEVICE_COMPILE__) && defined(__has_builtin)
#  if __has_builtin(__builtin_amdgcn_global_load_async_to_lds_b128)
#    define HAVE_ASYNC_LDS 1
#  endif
#endif
#ifndef HAVE_ASYNC_LDS
#  define HAVE_ASYNC_LDS 0
#endif

#if HAVE_ASYNC_LDS
typedef int v4i_vs __attribute__((vector_size(16)));               // matches builtin pointee
typedef __attribute__((address_space(1))) v4i_vs  glb_v4i;         // printed as "__device__" in HIP
typedef __attribute__((address_space(3))) v4i_vs  lds_v4i;         // printed as "__shared__" in HIP
__device__ __forceinline__ void async_b128(const void* g, void* l) {
    __builtin_amdgcn_global_load_async_to_lds_b128((glb_v4i*)g, (lds_v4i*)l, 0, 0);
}
#  if __has_builtin(__builtin_amdgcn_s_wait_asynccnt)
#    define ASYNC_WAIT() __builtin_amdgcn_s_wait_asynccnt(0)
#  else
#    define ASYNC_WAIT() asm volatile("s_wait_asynccnt 0x0" ::: "memory")
#  endif
#else
#  define ASYNC_WAIT() ((void)0)
#endif

__device__ __forceinline__ v8f wmma_bf16(const v16bf& a, const v16bf& b, const v8f& c) {
    return __builtin_amdgcn_wmma_f32_16x16x32_bf16(false, a, false, b, (short)0, c, false, false);
}

// ---------------------------------------------------------------------------
// Weight cast + transpose: out[n*K + k] = bf16(B[k][n]); headMode handles the
// (H, D, HS) layout of Wq/Wk/Wv so the fused N axis is h*64 + hs.
// ---------------------------------------------------------------------------
__global__ __launch_bounds__(256) void castT(const float* __restrict__ in,
                                             __bf16* __restrict__ out,
                                             int K, int N, int headMode) {
    size_t idx = (size_t)blockIdx.x * 256 + threadIdx.x;
    if (idx >= (size_t)K * N) return;
    int n = (int)(idx / K);
    int k = (int)(idx % K);
    float v = headMode ? in[((size_t)(n >> 6) * K + k) * 64 + (n & 63)]
                       : in[(size_t)k * N + n];
    out[idx] = (__bf16)v;
}

// ---------------------------------------------------------------------------
// LayerNorm over D=1024, one row per 256-thread block, bf16 output.
// ---------------------------------------------------------------------------
__global__ __launch_bounds__(256) void ln_bf16(const float* __restrict__ x,
                                               const float* __restrict__ g,
                                               const float* __restrict__ b,
                                               __bf16* __restrict__ out) {
    const int row  = blockIdx.x;
    const int t    = threadIdx.x;
    const int lane = t & 31, wave = t >> 5;
    const float* xr = x + (size_t)row * Dc;

    float vals[4], s = 0.f, s2 = 0.f;
#pragma unroll
    for (int i = 0; i < 4; ++i) {
        float v = xr[t + i * 256];
        vals[i] = v; s += v; s2 += v * v;
    }
#pragma unroll
    for (int off = 16; off >= 1; off >>= 1) {
        s  += __shfl_xor(s,  off, 32);
        s2 += __shfl_xor(s2, off, 32);
    }
    __shared__ float ws[8], ws2[8];
    if (lane == 0) { ws[wave] = s; ws2[wave] = s2; }
    __syncthreads();
    s = 0.f; s2 = 0.f;
#pragma unroll
    for (int w = 0; w < 8; ++w) { s += ws[w]; s2 += ws2[w]; }
    const float mu  = s * (1.0f / Dc);
    const float var = s2 * (1.0f / Dc) - mu * mu;
    const float rs  = rsqrtf(var + 1e-5f);
#pragma unroll
    for (int i = 0; i < 4; ++i) {
        int c = t + i * 256;
        out[(size_t)row * Dc + c] = (__bf16)((vals[i] - mu) * rs * g[c] + b[c]);
    }
}

// ---------------------------------------------------------------------------
// bf16 GEMM: C[M,N] = A[M,K] * B[K,N], Bt = B^T ([N][K] row-major).
// 256 threads = 8 waves (4x2); block tile 128x128; wave tile 32x64;
// K step 32; double-buffered LDS tiles staged with async-LDS loads.
// Epilogue MODE: 0 = Q/K scatter [B,H,T,HS]; 1 = V^T scatter [B,H,HS,T];
//               2/4 = bias + f32 residual -> f32 out; 3 = bias + ReLU -> bf16.
// ---------------------------------------------------------------------------
template <int MODE>
__global__ __launch_bounds__(256) void gemm_bf16_k(const __bf16* __restrict__ A,
                                                   const __bf16* __restrict__ Bt,
                                                   int K, int N,
                                                   const float* __restrict__ bias,
                                                   const float* __restrict__ res,
                                                   float* __restrict__ outF,
                                                   __bf16* __restrict__ outB) {
    __shared__ __align__(16) __bf16 Asm[2][128 * 32];
    __shared__ __align__(16) __bf16 Bsm[2][128 * 32];

    const int t      = threadIdx.x;
    const int lane   = t & 31;
    const int wave   = t >> 5;
    const int lanelo = lane & 15;
    const int laneHi = lane >> 4;         // 0 or 1
    const int wm     = wave >> 1;         // 0..3  (M sub-tile of 32)
    const int wn     = wave & 1;          // 0..1  (N sub-tile of 64)
    const int bm     = blockIdx.y * 128;
    const int bn     = blockIdx.x * 128;

    const int ldRow  = t >> 1;            // 0..127
    const int ldHalf = (t & 1) * 16;      // element offset in 32-wide K row

    const __bf16* gA = A  + (size_t)(bm + ldRow) * K + ldHalf;
    const __bf16* gB = Bt + (size_t)(bn + ldRow) * K + ldHalf;

    auto stage = [&](int buf, int k0) {
        const __bf16* ga = gA + k0;
        const __bf16* gb = gB + k0;
        __bf16* la = &Asm[buf][ldRow * 32 + ldHalf];
        __bf16* lb = &Bsm[buf][ldRow * 32 + ldHalf];
#if HAVE_ASYNC_LDS
        async_b128(ga,     la);
        async_b128(ga + 8, la + 8);
        async_b128(gb,     lb);
        async_b128(gb + 8, lb + 8);
#else
        ((uint4*)la)[0] = ((const uint4*)ga)[0];
        ((uint4*)la)[1] = ((const uint4*)ga)[1];
        ((uint4*)lb)[0] = ((const uint4*)gb)[0];
        ((uint4*)lb)[1] = ((const uint4*)gb)[1];
#endif
    };

    v8f acc[2][4];
#pragma unroll
    for (int mi = 0; mi < 2; ++mi)
#pragma unroll
        for (int ni = 0; ni < 4; ++ni) acc[mi][ni] = (v8f){};

    stage(0, 0);
    ASYNC_WAIT();
    __syncthreads();

    int buf = 0;
    for (int k0 = 0; k0 < K; k0 += 32) {
        if (k0 + 32 < K) stage(buf ^ 1, k0 + 32);   // DMA next tile during compute

        FragBF af[2], bf[4];
#pragma unroll
        for (int mi = 0; mi < 2; ++mi) {
            int base = (wm * 32 + mi * 16 + lanelo) * 32 + (laneHi ? 8 : 0);
            af[mi].u[0] = *(const uint4*)&Asm[buf][base];
            af[mi].u[1] = *(const uint4*)&Asm[buf][base + 16];
        }
#pragma unroll
        for (int ni = 0; ni < 4; ++ni) {
            int base = (wn * 64 + ni * 16 + lanelo) * 32 + (laneHi ? 8 : 0);
            bf[ni].u[0] = *(const uint4*)&Bsm[buf][base];
            bf[ni].u[1] = *(const uint4*)&Bsm[buf][base + 16];
        }
#pragma unroll
        for (int mi = 0; mi < 2; ++mi)
#pragma unroll
            for (int ni = 0; ni < 4; ++ni)
                acc[mi][ni] = wmma_bf16(af[mi].v, bf[ni].v, acc[mi][ni]);

        ASYNC_WAIT();
        __syncthreads();
        buf ^= 1;
    }

    // epilogue: C layout per set: row = g + 8*laneHi, col = lanelo
#pragma unroll
    for (int mi = 0; mi < 2; ++mi) {
#pragma unroll
        for (int ni = 0; ni < 4; ++ni) {
#pragma unroll
            for (int g = 0; g < 8; ++g) {
                const int r = bm + wm * 32 + mi * 16 + g + 8 * laneHi;
                const int c = bn + wn * 64 + ni * 16 + lanelo;
                float v = acc[mi][ni][g];
                if constexpr (MODE == 0) {          // Q/K -> [B,H,T,HS] bf16
                    int bI = r >> 11, tt = r & (Tc - 1);
                    int h  = c >> 6,  hs = c & 63;
                    outB[(((size_t)bI * Hc + h) * Tc + tt) * HSc + hs] = (__bf16)v;
                } else if constexpr (MODE == 1) {   // V -> [B,H,HS,T] bf16
                    int bI = r >> 11, tt = r & (Tc - 1);
                    int h  = c >> 6,  hs = c & 63;
                    outB[(((size_t)bI * Hc + h) * HSc + hs) * Tc + tt] = (__bf16)v;
                } else if constexpr (MODE == 2 || MODE == 4) { // bias+residual, f32 out
                    outF[(size_t)r * N + c] = v + bias[c] + res[(size_t)r * N + c];
                } else if constexpr (MODE == 3) {   // bias + ReLU, bf16 out
                    float z = v + bias[c];
                    outB[(size_t)r * N + c] = (__bf16)(z > 0.f ? z : 0.f);
                }
            }
        }
    }
}

// ---------------------------------------------------------------------------
// Flash attention, causal. One block per (64-query tile, head, batch).
// 128 threads = 4 waves; wave owns 16 query rows. QK^T and PV via WMMA.
// V arrives pre-transposed [B,H,HS,T] so PV B-fragments are contiguous.
// K/V tiles double-buffered in LDS; next tile streamed via async-LDS loads
// while the current tile's S and PV WMMAs run.
// ---------------------------------------------------------------------------
__global__ __launch_bounds__(128) void attn_kernel(const __bf16* __restrict__ Q,
                                                   const __bf16* __restrict__ Kg,
                                                   const __bf16* __restrict__ Vt,
                                                   __bf16* __restrict__ attn) {
    __shared__ __align__(16) __bf16 Ksm[2][64 * 64];
    __shared__ __align__(16) __bf16 Vsm[2][64 * 64];
    __shared__ __align__(16) __bf16 Psm[4][16 * 64];

    const int qt = blockIdx.x, h = blockIdx.y, b = blockIdx.z;
    const int bh = b * Hc + h;
    const int t = threadIdx.x;
    const int lane = t & 31, wave = t >> 5;
    const int lanelo = lane & 15, laneHi = lane >> 4;

    const int ldRow  = t >> 1;           // 0..63
    const int ldHalf = (t & 1) * 32;     // elements (64 bytes)

    const __bf16* gK = Kg + ((size_t)bh * Tc + ldRow) * HSc + ldHalf;
    const __bf16* gV = Vt + ((size_t)bh * HSc + ldRow) * Tc + ldHalf;

    auto stage = [&](int buf, int kt) {
        const __bf16* kp = gK + (size_t)kt * 64 * HSc;   // advance 64 key rows
        const __bf16* vp = gV + kt * 64;                 // advance 64 columns
        __bf16* kd = &Ksm[buf][ldRow * 64 + ldHalf];
        __bf16* vd = &Vsm[buf][ldRow * 64 + ldHalf];
#if HAVE_ASYNC_LDS
#pragma unroll
        for (int j = 0; j < 4; ++j) {
            async_b128(kp + 8 * j, kd + 8 * j);
            async_b128(vp + 8 * j, vd + 8 * j);
        }
#else
#pragma unroll
        for (int j = 0; j < 4; ++j) {
            ((uint4*)kd)[j] = ((const uint4*)kp)[j];
            ((uint4*)vd)[j] = ((const uint4*)vp)[j];
        }
#endif
    };

    // Q A-fragments for this wave's 16 rows (two 32-wide K chunks of HS=64)
    FragBF qf[2];
    {
        const int qrow = qt * 64 + wave * 16 + lanelo;
        const __bf16* qp = Q + ((size_t)bh * Tc + qrow) * HSc;
#pragma unroll
        for (int c = 0; c < 2; ++c) {
            int off = c * 32 + (laneHi ? 8 : 0);
            qf[c].u[0] = *(const uint4*)(qp + off);
            qf[c].u[1] = *(const uint4*)(qp + off + 16);
        }
    }

    v8f oacc[4];
#pragma unroll
    for (int ni = 0; ni < 4; ++ni) oacc[ni] = (v8f){};
    float mrow[8], lrow[8];
#pragma unroll
    for (int g = 0; g < 8; ++g) { mrow[g] = -3.0e38f; lrow[g] = 0.f; }

    stage(0, 0);
    ASYNC_WAIT();
    __syncthreads();

    int buf = 0;
    for (int kt = 0; kt <= qt; ++kt) {
        if (kt + 1 <= qt) stage(buf ^ 1, kt + 1);   // DMA next K/V tile

        // S = Q * K^T  (16x64 per wave, 4 N-tiles, 2 K-chunks)
        v8f sacc[4];
#pragma unroll
        for (int ni = 0; ni < 4; ++ni) sacc[ni] = (v8f){};
#pragma unroll
        for (int c = 0; c < 2; ++c) {
#pragma unroll
            for (int ni = 0; ni < 4; ++ni) {
                FragBF kb;
                int base = (ni * 16 + lanelo) * 64 + c * 32 + (laneHi ? 8 : 0);
                kb.u[0] = *(const uint4*)&Ksm[buf][base];
                kb.u[1] = *(const uint4*)&Ksm[buf][base + 16];
                sacc[ni] = wmma_bf16(qf[c].v, kb.v, sacc[ni]);
            }
        }

        // scale + causal mask + online softmax
        const bool diag = (kt == qt);
        float mnew[8], rsum[8];
#pragma unroll
        for (int g = 0; g < 8; ++g) {
            const int rloc = wave * 16 + g + 8 * laneHi;  // row in 64-q tile
            float rmax = -3.0e38f;
#pragma unroll
            for (int ni = 0; ni < 4; ++ni) {
                float s = sacc[ni][g] * 0.125f;           // 1/sqrt(64)
                if (diag && (ni * 16 + lanelo) > rloc) s = -3.0e38f;
                sacc[ni][g] = s;
                rmax = fmaxf(rmax, s);
            }
#pragma unroll
            for (int off = 1; off <= 8; off <<= 1)
                rmax = fmaxf(rmax, __shfl_xor(rmax, off, 32));
            mnew[g] = fmaxf(mrow[g], rmax);
            float rs = 0.f;
#pragma unroll
            for (int ni = 0; ni < 4; ++ni) {
                float p = __expf(sacc[ni][g] - mnew[g]);
                rs += p;
                Psm[wave][(g + 8 * laneHi) * 64 + ni * 16 + lanelo] = (__bf16)p;
            }
#pragma unroll
            for (int off = 1; off <= 8; off <<= 1)
                rs += __shfl_xor(rs, off, 32);
            rsum[g] = rs;
        }
#pragma unroll
        for (int g = 0; g < 8; ++g) {
            float sc = __expf(mrow[g] - mnew[g]);
            lrow[g] = lrow[g] * sc + rsum[g];
            mrow[g] = mnew[g];
#pragma unroll
            for (int ni = 0; ni < 4; ++ni) oacc[ni][g] *= sc;
        }
        __syncthreads();  // Psm visible (cross-lane), Vsm[buf] stable

        // O += P * V  (A = P 16x64 via LDS, B = V^T rows)
#pragma unroll
        for (int c = 0; c < 2; ++c) {
            FragBF pf;
            int pbase = lanelo * 64 + c * 32 + (laneHi ? 8 : 0);
            pf.u[0] = *(const uint4*)&Psm[wave][pbase];
            pf.u[1] = *(const uint4*)&Psm[wave][pbase + 16];
#pragma unroll
            for (int ni = 0; ni < 4; ++ni) {
                FragBF vf;
                int base = (ni * 16 + lanelo) * 64 + c * 32 + (laneHi ? 8 : 0);
                vf.u[0] = *(const uint4*)&Vsm[buf][base];
                vf.u[1] = *(const uint4*)&Vsm[buf][base + 16];
                oacc[ni] = wmma_bf16(pf.v, vf.v, oacc[ni]);
            }
        }

        ASYNC_WAIT();      // next tile landed in LDS
        __syncthreads();   // publish; all reads of old buffers complete
        buf ^= 1;
    }

    // normalize + write concat-head output [B,T,D] bf16
#pragma unroll
    for (int g = 0; g < 8; ++g) {
        const float inv = 1.0f / lrow[g];
        const int trow = qt * 64 + wave * 16 + g + 8 * laneHi;
#pragma unroll
        for (int ni = 0; ni < 4; ++ni) {
            const int col = h * HSc + ni * 16 + lanelo;
            attn[((size_t)b * Tc + trow) * Dc + col] = (__bf16)(oacc[ni][g] * inv);
        }
    }
}

// ---------------------------------------------------------------------------
extern "C" void kernel_launch(void* const* d_in, const int* in_sizes, int n_in,
                              void* d_out, int out_size, void* d_ws, size_t ws_size,
                              hipStream_t stream) {
    (void)in_sizes; (void)n_in; (void)out_size; (void)ws_size;
    const float* x     = (const float*)d_in[0];
    const float* Wq    = (const float*)d_in[1];
    const float* Wk    = (const float*)d_in[2];
    const float* Wv    = (const float*)d_in[3];
    const float* Wproj = (const float*)d_in[4];
    const float* bproj = (const float*)d_in[5];
    const float* ln1g  = (const float*)d_in[6];
    const float* ln1b  = (const float*)d_in[7];
    const float* ln2g  = (const float*)d_in[8];
    const float* ln2b  = (const float*)d_in[9];
    const float* W1    = (const float*)d_in[10];
    const float* b1    = (const float*)d_in[11];
    const float* W2    = (const float*)d_in[12];
    const float* b2    = (const float*)d_in[13];

    char* p = (char*)d_ws;
    auto take = [&](size_t bytes) { void* r = p; p += (bytes + 255) & ~(size_t)255; return r; };

    const size_t BTD = (size_t)Mrows * Dc;        // 8M elements
    __bf16* hb   = (__bf16*)take(BTD * 2);
    __bf16* WtQ  = (__bf16*)take((size_t)Dc * Dc * 2);
    __bf16* WtK  = (__bf16*)take((size_t)Dc * Dc * 2);
    __bf16* WtV  = (__bf16*)take((size_t)Dc * Dc * 2);
    __bf16* WtP  = (__bf16*)take((size_t)Dc * Dc * 2);
    __bf16* Wt1  = (__bf16*)take((size_t)4 * Dc * Dc * 2);
    __bf16* Wt2  = (__bf16*)take((size_t)4 * Dc * Dc * 2);
    __bf16* Qb   = (__bf16*)take(BTD * 2);
    __bf16* Kb   = (__bf16*)take(BTD * 2);
    __bf16* Vtb  = (__bf16*)take(BTD * 2);
    __bf16* attb = (__bf16*)take(BTD * 2);
    float*  x2   = (float*) take(BTD * 4);
    __bf16* h2   = (__bf16*)take(BTD * 2);
    __bf16* act1 = (__bf16*)take((size_t)Mrows * 4 * Dc * 2);

    const int g1M = (int)((size_t)Dc * Dc / 256);        // 4096 blocks
    const int g4M = (int)((size_t)4 * Dc * Dc / 256);
    castT<<<g1M, 256, 0, stream>>>(Wq,    WtQ, Dc, Dc,     1);
    castT<<<g1M, 256, 0, stream>>>(Wk,    WtK, Dc, Dc,     1);
    castT<<<g1M, 256, 0, stream>>>(Wv,    WtV, Dc, Dc,     1);
    castT<<<g1M, 256, 0, stream>>>(Wproj, WtP, Dc, Dc,     0);
    castT<<<g4M, 256, 0, stream>>>(W1,    Wt1, Dc, 4 * Dc, 0);
    castT<<<g4M, 256, 0, stream>>>(W2,    Wt2, 4 * Dc, Dc, 0);

    ln_bf16<<<Mrows, 256, 0, stream>>>(x, ln1g, ln1b, hb);

    dim3 gD(Dc / 128, Mrows / 128);            // (8, 64)
    dim3 g4D(4 * Dc / 128, Mrows / 128);       // (32, 64)
    gemm_bf16_k<0><<<gD, 256, 0, stream>>>(hb, WtQ, Dc, Dc, nullptr, nullptr, nullptr, Qb);
    gemm_bf16_k<0><<<gD, 256, 0, stream>>>(hb, WtK, Dc, Dc, nullptr, nullptr, nullptr, Kb);
    gemm_bf16_k<1><<<gD, 256, 0, stream>>>(hb, WtV, Dc, Dc, nullptr, nullptr, nullptr, Vtb);

    attn_kernel<<<dim3(Tc / 64, Hc, Bc), 128, 0, stream>>>(Qb, Kb, Vtb, attb);

    gemm_bf16_k<2><<<gD, 256, 0, stream>>>(attb, WtP, Dc, Dc, bproj, x, x2, nullptr);
    ln_bf16<<<Mrows, 256, 0, stream>>>(x2, ln2g, ln2b, h2);
    gemm_bf16_k<3><<<g4D, 256, 0, stream>>>(h2, Wt1, Dc, 4 * Dc, b1, nullptr, nullptr, act1);
    gemm_bf16_k<4><<<gD, 256, 0, stream>>>(act1, Wt2, 4 * Dc, Dc, b2, x2, (float*)d_out, nullptr);
}
